// AttentionValueIteration_31224412242000
// MI455X (gfx1250) — matrix operand order repaired and checked by reading, not compiled, not run on hardware
//
#include <hip/hip_runtime.h>
#include <hip/hip_bf16.h>

// ---------------------------------------------------------------------------
// AttentionValueIteration fused kernel for gfx1250 (MI455X, wave32, WMMA).
//
// Static shapes: B=8, P=1, A=8, D=8, K=3, T=H=W=32.  PA=8, PA*D=64.
// qk conv == GEMM [voxels x 27taps] x [27taps x 128ch]; v conv appended as
// 8 extra columns (softmaxed weights) -> one [32 x 144] f16 weight matrix,
// consumed as 9 WMMA B fragments of 32x16 (V_WMMA_F32_16X16X32_F16).
// ---------------------------------------------------------------------------

typedef __attribute__((ext_vector_type(16))) _Float16 v16h;
typedef __attribute__((ext_vector_type(8)))  _Float16 v8h;
typedef __attribute__((ext_vector_type(8)))  float    v8f;

#define NTAP   27
#define NFRAG  9        // 144 channels / 16
#define EXP_N  216      // 6*6*6 expanded voxels per block
#define EXP_P  224      // padded to 14 WMMA tiles (rows 216..223 are scratch)
#define NTILE  14

// tap K -> offset into the 8x8x8 x-tile (compile-time folded under unroll)
__device__ constexpr int tapoff(int K) {
    return (K / 9) * 64 + ((K / 3) % 3) * 8 + (K % 3);
}

// ---------------------------------------------------------------------------
// prep_w: softmax the value-conv weights and emit the combined weight matrix
// in exact WMMA B-fragment layout (lane l: N = l&15; element i: K = (i<8 ? i : i+8)+8*(l>>4)).
// wsB[f*512 + l*16 + i]  (f16)
// ---------------------------------------------------------------------------
__global__ __launch_bounds__(256) void avi_prep_w(const float* __restrict__ w_qk,
                                                  const float* __restrict__ w_v,
                                                  _Float16* __restrict__ wsB) {
    __shared__ float smv[8 * NTAP];
    const int tid = threadIdx.x;
    if (tid < 8) {
        float mx = -1e30f;
        for (int t = 0; t < NTAP; ++t) mx = fmaxf(mx, w_v[tid * NTAP + t]);
        float s = 0.f;
        for (int t = 0; t < NTAP; ++t) s += __expf(w_v[tid * NTAP + t] - mx);
        const float inv = 1.f / s;
        for (int t = 0; t < NTAP; ++t)
            smv[tid * NTAP + t] = __expf(w_v[tid * NTAP + t] - mx) * inv;
    }
    __syncthreads();
    for (int idx = tid; idx < NFRAG * 512; idx += 256) {
        const int f = idx >> 9;
        const int rem = idx & 511;
        const int l = rem >> 4;
        const int i = rem & 15;
        const int K = (i < 8 ? i : i + 8) + 8 * (l >> 4);       // B layout: K per lane-half
        const int c = f * 16 + (l & 15);       // output channel (N)
        float val = 0.f;
        if (K < NTAP) {
            if (c < 128)       val = w_qk[c * NTAP + K];          // q,k conv weights
            else if (c < 136)  val = smv[(c - 128) * NTAP + K];   // softmaxed v weights
        }
        wsB[idx] = (_Float16)val;
    }
}

// ---------------------------------------------------------------------------
// avi_main: one block = one 4x4x4 interior tile of one batch. 256 thr = 8 waves.
// ---------------------------------------------------------------------------
__global__ __launch_bounds__(256) void avi_main(const float* __restrict__ values,
                                                const float* __restrict__ rewards,
                                                const _Float16* __restrict__ wsB,
                                                float* __restrict__ out) {
    // Manually packed LDS (61.5 KB).  The first 2 KB region holds the x halo
    // tile during stages 1-2 and is reused for q-values in stages 3-4 (the
    // live ranges are separated by a barrier).
    __shared__ __align__(16) unsigned char smem[2048 + EXP_P * 64 * 2 * 2 + EXP_P * 8 * 2];
    float*    const xt    = (float*)smem;                       // [512]
    float*    const qvals = (float*)smem;                       // [512] (alias)
    _Float16* const qlds  = (_Float16*)(smem + 2048);           // [224*64]
    _Float16* const klds  = (_Float16*)(smem + 2048 + 28672);   // [224*64]
    _Float16* const vlds  = (_Float16*)(smem + 2048 + 57344);   // [224*8]

    const int bid = blockIdx.x;
    const int bw = bid & 7, bh = (bid >> 3) & 7, bt = (bid >> 6) & 7, b = bid >> 9;
    const int ot = bt * 4, oh = bh * 4, ow = bw * 4;
    const int tid = threadIdx.x;
    const int lane = tid & 31, wave = tid >> 5;
    const int half = lane >> 4, col = lane & 15;

    // ---- stage 1: x = values + rewards into LDS (zero outside [0,32)^3) ----
    for (int i = tid; i < 512; i += 256) {
        const int lt = i >> 6, lh = (i >> 3) & 7, lw = i & 7;
        const int gt = ot + lt - 2, gh = oh + lh - 2, gw = ow + lw - 2;
        float x = 0.f;
        if ((unsigned)gt < 32u && (unsigned)gh < 32u && (unsigned)gw < 32u) {
            const int gi = ((b * 32 + gt) * 32 + gh) * 32 + gw;
            x = values[gi] + rewards[gi];
        }
        xt[i] = x;
    }

    // ---- weight B fragments (coalesced v16h per lane, shared by all tiles) ----
    v16h bfr[NFRAG];
    const v16h* wb = (const v16h*)wsB;
#pragma unroll
    for (int f = 0; f < NFRAG; ++f) bfr[f] = wb[f * 32 + lane];

    __syncthreads();

    // ---- stage 2: q,k,v via WMMA over expanded (padded) 224 voxel rows ----
    // Tile loop is wave-uniform -> EXEC is all-ones at every WMMA.
    for (int tl = wave; tl < NTILE; tl += 8) {
        // ---- A fragment gather: M = lane&15, K per ISA 16-bit A layout. ----
        const int mrow = tl * 16 + col;
        const int evc = mrow < EXP_N ? mrow : 0;
        const int et = evc / 36, eh = (evc / 6) % 6, ew = evc % 6;
        const int gbase = (et << 6) + (eh << 3) + ew;

        float xv[16];
#pragma unroll
        for (int i = 0; i < 16; ++i) {
            // half=0 lanes hold K = {0..7,16..23}; half=1 hold K = {8..15,24..31}
            const int K0 = (i < 8) ? i : i + 8;
            const int K1 = (i < 8) ? i + 8 : i + 16;
            const int o0 = tapoff(K0);
            const int o1 = (K1 < NTAP) ? tapoff(K1) : 0;   // safe dummy addr
            const int off = half ? o1 : o0;                // one v_cndmask
            xv[i] = xt[gbase + off];                       // independent ds_loads
        }
        v16h a;
#pragma unroll
        for (int i = 0; i < 16; ++i) {
            const int K1 = (i < 8) ? i + 8 : i + 16;
            float x = xv[i];
            if (K1 >= NTAP) x = half ? 0.f : x;            // zero pad taps 27..31
            a[i] = (_Float16)x;
        }

        // ---- per-row decode + in-grid mask: ONCE per tile (8 rows/lane) ----
        const int rowbase = tl * 16 + 8 * half;
        float mrow_mask[8];
        int   rowoff[8];
#pragma unroll
        for (int r = 0; r < 8; ++r) {
            const int ev2 = rowbase + r;                   // may be >=216: scratch row
            const int et2 = ev2 / 36;
            const int rm  = ev2 - et2 * 36;
            const int eh2 = rm / 6;
            const int ew2 = rm - eh2 * 6;
            const int pt = ot + et2 - 1, ph = oh + eh2 - 1, pw = ow + ew2 - 1;
            const bool inGrid = (unsigned)pt < 32u && (unsigned)ph < 32u &&
                                (unsigned)pw < 32u;
            mrow_mask[r] = inGrid ? 1.f : 0.f;             // zero k/v outside grid
            rowoff[r] = ev2 * 64;
        }

        // ---- all 9 WMMAs first, with INDEPENDENT accumulators, so they can
        //      issue back-to-back (no WAW serialization, hazard slots filled
        //      by the store-side VALU of earlier fragments). ----
        const v8f zero = {};
        v8f dfr[NFRAG];
#pragma unroll
        for (int f = 0; f < NFRAG; ++f)
            dfr[f] = __builtin_amdgcn_wmma_f32_16x16x32_f16(
                false, a, false, bfr[f], (short)0, zero, false, false);

#pragma unroll
        for (int f = 0; f < NFRAG; ++f) {
            if (f < 4) {                                   // q channels 0..63
#pragma unroll
                for (int r = 0; r < 8; ++r)
                    qlds[rowoff[r] + f * 16 + col] = (_Float16)(dfr[f][r] * mrow_mask[r]);
            } else if (f < 8) {                            // k channels 64..127
#pragma unroll
                for (int r = 0; r < 8; ++r)
                    klds[rowoff[r] + (f - 4) * 16 + col] = (_Float16)(dfr[f][r] * mrow_mask[r]);
            } else {                                       // v channels 128..135
                if (col < 8) {
#pragma unroll
                    for (int r = 0; r < 8; ++r)
                        vlds[(rowbase + r) * 8 + col] = (_Float16)(dfr[f][r] * mrow_mask[r]);
                }
            }
        }
    }
    __syncthreads();

    // ---- stage 3: 3x3x3 neighborhood attention, online softmax over 27 taps ----
    for (int p = tid; p < 512; p += 256) {
        const int vox = p & 63;
        const int act = p >> 6;
        const int it = vox >> 4, ih = (vox >> 2) & 3, iw = vox & 3;
        const int evq = (it + 1) * 36 + (ih + 1) * 6 + (iw + 1);
        const v8h q8 = *(const v8h*)(qlds + evq * 64 + act * 8);
        float qf[8];
#pragma unroll
        for (int d = 0; d < 8; ++d) qf[d] = (float)q8[d];

        float m = -1e30f, den = 0.f, num = 0.f;
#pragma unroll
        for (int kt = 0; kt < 3; ++kt)
#pragma unroll
            for (int kh = 0; kh < 3; ++kh)
#pragma unroll
                for (int kw = 0; kw < 3; ++kw) {
                    const int ev = (it + kt) * 36 + (ih + kh) * 6 + (iw + kw);
                    const v8h k8 = *(const v8h*)(klds + ev * 64 + act * 8);
                    float s = 0.f;
#pragma unroll
                    for (int d = 0; d < 8; ++d) s += qf[d] * (float)k8[d];
                    const float vn = (float)vlds[ev * 8 + act];
                    const float nm = fmaxf(m, s);
                    const float cs = __expf(m - nm);       // rescale old accumulators
                    const float e  = __expf(s - nm);
                    den = den * cs + e;
                    num = num * cs + e * vn;
                    m = nm;
                }
        qvals[act * 64 + vox] = num / den;
    }
    __syncthreads();

    // ---- stage 4: hard max over actions, write output ----
    if (tid < 64) {
        float mv = qvals[tid];
#pragma unroll
        for (int a2 = 1; a2 < 8; ++a2) mv = fmaxf(mv, qvals[a2 * 64 + tid]);
        const int it = tid >> 4, ih = (tid >> 2) & 3, iw = tid & 3;
        out[((b * 32 + ot + it) * 32 + oh + ih) * 32 + ow + iw] = mv;
    }
}

// ---------------------------------------------------------------------------
extern "C" void kernel_launch(void* const* d_in, const int* in_sizes, int n_in,
                              void* d_out, int out_size, void* d_ws, size_t ws_size,
                              hipStream_t stream) {
    (void)in_sizes; (void)n_in; (void)out_size; (void)ws_size;
    const float* values  = (const float*)d_in[0];
    const float* rewards = (const float*)d_in[1];
    const float* w_qk    = (const float*)d_in[2];
    const float* w_v     = (const float*)d_in[3];
    float* out = (float*)d_out;
    _Float16* wsB = (_Float16*)d_ws;   // 9*512 f16 = 9216 bytes

    avi_prep_w<<<1, 256, 0, stream>>>(w_qk, w_v, wsB);
    avi_main<<<4096, 256, 0, stream>>>(values, rewards, wsB, out);
}
